// ArcMarginProduct_virface_19877108646226
// MI455X (gfx1250) — matrix operand order repaired and checked by reading, not compiled, not run on hardware
//
#include <hip/hip_runtime.h>
#include <stdint.h>

// ---------- problem constants ----------
#define B_M 512
#define D_K 512
#define C_N 84281

// ---------- GEMM tiling ----------
#define BM 128
#define BN 128
#define BK 32
#define STAGES (D_K / BK)   // 16
#define FRAG_STRIDE 34      // f16 elems per LDS frag row (32 + 2 pad: conflict-free gathers)

typedef __attribute__((ext_vector_type(16))) _Float16 v16h;
typedef __attribute__((ext_vector_type(8)))  float    v8f;

union FragH { v16h v; unsigned u[8]; };

// LDS byte-offset of a __shared__ object (generic -> addrspace(3) -> 32-bit offset)
typedef __attribute__((address_space(3))) const char lds_cchar;
__device__ __forceinline__ uint32_t ldsOff(const void* p) {
  return (uint32_t)(uintptr_t)(lds_cchar*)p;
}

// CDNA5 async copy: global -> LDS, 16B per lane, tracked by ASYNCcnt
__device__ __forceinline__ void async_copy_b128(uint32_t lds_byte_off, const void* gaddr) {
  asm volatile("global_load_async_to_lds_b128 %0, %1, off"
               :: "v"(lds_byte_off), "v"(gaddr) : "memory");
}
template<int N>
__device__ __forceinline__ void waitAsync() {
  asm volatile("s_wait_asynccnt %0" :: "n"(N) : "memory");
}

// pack two f16 into one dword
__device__ __forceinline__ unsigned packh(_Float16 a, _Float16 b) {
  unsigned short ua = __builtin_bit_cast(unsigned short, a);
  unsigned short ub = __builtin_bit_cast(unsigned short, b);
  return (unsigned)ua | ((unsigned)ub << 16);
}

// split float4 into f16 hi/lo and store 4+4 elems (two dwords each) to LDS
__device__ __forceinline__ void cvtStore(unsigned short* hi, unsigned short* lo, float4 f) {
  _Float16 h0 = (_Float16)f.x, h1 = (_Float16)f.y, h2 = (_Float16)f.z, h3 = (_Float16)f.w;
  _Float16 l0 = (_Float16)(f.x - (float)h0);
  _Float16 l1 = (_Float16)(f.y - (float)h1);
  _Float16 l2 = (_Float16)(f.z - (float)h2);
  _Float16 l3 = (_Float16)(f.w - (float)h3);
  ((unsigned*)hi)[0] = packh(h0, h1);
  ((unsigned*)hi)[1] = packh(h2, h3);
  ((unsigned*)lo)[0] = packh(l0, l1);
  ((unsigned*)lo)[1] = packh(l2, l3);
}

// ---------- kernel 1: per-row inverse L2 norm (one wave32 per row) ----------
__global__ void __launch_bounds__(256) rnorm_kernel(const float* __restrict__ src,
                                                    float* __restrict__ dst, int rows) {
  int wid  = (int)(blockIdx.x * 256 + threadIdx.x) >> 5;
  int lane = threadIdx.x & 31;
  if (wid >= rows) return;
  const float4* r4 = (const float4*)(src + (size_t)wid * D_K);
  float s = 0.0f;
#pragma unroll
  for (int j = 0; j < 4; ++j) {
    float4 f = r4[lane + 32 * j];
    s += f.x * f.x + f.y * f.y + f.z * f.z + f.w * f.w;
  }
#pragma unroll
  for (int off = 16; off; off >>= 1) s += __shfl_xor(s, off, 32);
  if (lane == 0) dst[wid] = 1.0f / fmaxf(sqrtf(s), 1e-12f);
}

// ---------- kernel 2: fused f16x3 WMMA GEMM + ArcFace margin epilogue ----------
__global__ void __launch_bounds__(256) arc_gemm(const float* __restrict__ X,
                                                const float* __restrict__ W,
                                                const long long* __restrict__ label,
                                                const float* __restrict__ rX,
                                                const float* __restrict__ rW,
                                                float* __restrict__ out) {
  __shared__ __align__(16) float Afp[2][BM * BK];           // 32 KB, async landing (double buf)
  __shared__ __align__(16) float Bfp[2][BN * BK];           // 32 KB
  __shared__ __align__(8) unsigned short Ahi[BM * FRAG_STRIDE];  // 8.5 KB each
  __shared__ __align__(8) unsigned short Alo[BM * FRAG_STRIDE];
  __shared__ __align__(8) unsigned short Bhi[BN * FRAG_STRIDE];
  __shared__ __align__(8) unsigned short Blo[BN * FRAG_STRIDE];
  __shared__ float rnA[BM];
  __shared__ float rnW[BN];
  __shared__ int   lab[BM];

  const int t  = threadIdx.x;
  const int m0 = blockIdx.y * BM;
  const int n0 = blockIdx.x * BN;

  if (t < BM) {
    rnA[t] = rX[m0 + t];
    lab[t] = (int)label[m0 + t];
    int nn = n0 + t; if (nn > C_N - 1) nn = C_N - 1;
    rnW[t] = rW[nn];
  }

  const int lane = t & 31;
  const int w    = t >> 5;
  const int wm   = w & 3;     // M quadrant (4 x 32 rows)
  const int wn   = w >> 2;    // N half (2 x 64 cols)
  const int h    = lane >> 4; // half-wave
  const int ml   = lane & 15;

  v8f acc[2][4] = {};

  // ---- async stage issue: 1024 16B chunks per matrix per stage, 4+4 per thread ----
  auto issueStage = [&](int stage) {
    const int kk  = stage * BK;
    const int buf = stage & 1;
    const uint32_t aOff = ldsOff(&Afp[buf][0]);
    const uint32_t bOff = ldsOff(&Bfp[buf][0]);
#pragma unroll
    for (int j = 0; j < 4; ++j) {
      int q = t + 256 * j;          // chunk id 0..1023
      int row = q >> 3, c = q & 7;  // 8 chunks per 32-float row
      async_copy_b128(aOff + q * 16, X + (size_t)(m0 + row) * D_K + kk + c * 4);
      int rw_ = n0 + row; if (rw_ > C_N - 1) rw_ = C_N - 1;  // clamp OOB weight rows
      async_copy_b128(bOff + q * 16, W + (size_t)rw_ * D_K + kk + c * 4);
    }
  };

  // ---- cooperative fp32 -> f16 hi/lo conversion into padded frag tiles ----
  auto convert = [&](int buf) {
#pragma unroll
    for (int j = 0; j < 4; ++j) {
      int e4 = t + 256 * j;         // float4 id 0..1023
      int e = e4 * 4;
      int row = e >> 5, k = e & 31;
      float4 fa = ((const float4*)&Afp[buf][0])[e4];
      float4 fb = ((const float4*)&Bfp[buf][0])[e4];
      cvtStore(&Ahi[row * FRAG_STRIDE + k], &Alo[row * FRAG_STRIDE + k], fa);
      cvtStore(&Bhi[row * FRAG_STRIDE + k], &Blo[row * FRAG_STRIDE + k], fb);
    }
  };

  issueStage(0);
  for (int i = 0; i < STAGES; ++i) {
    if (i + 1 < STAGES) { issueStage(i + 1); waitAsync<8>(); }
    else                { waitAsync<0>(); }
    __syncthreads();                 // stage i fp32 tiles visible to all waves
    convert(i & 1);
    __syncthreads();                 // f16 hi/lo tiles ready

    // A fragments: 16-bit A 16x32 layout -> k0 = 2*(v%4) + 16*(v/4) + 8*h
    FragH a_hi[2], a_lo[2];
#pragma unroll
    for (int s = 0; s < 2; ++s) {
      int mrow = wm * 32 + s * 16 + ml;
      const unsigned* ha = (const unsigned*)&Ahi[mrow * FRAG_STRIDE];
      const unsigned* la = (const unsigned*)&Alo[mrow * FRAG_STRIDE];
#pragma unroll
      for (int v = 0; v < 8; ++v) {
        int k0 = ((v & 3) << 1) + ((v >> 2) << 4) + (h << 3);
        a_hi[s].u[v] = ha[k0 >> 1];
        a_lo[s].u[v] = la[k0 >> 1];
      }
    }
#pragma unroll
    for (int tb = 0; tb < 4; ++tb) {
      // B fragments: 16-bit B 32x16 layout -> k0 = 2*(v%4) + 16*(v/4) + 8*h
      FragH b_hi, b_lo;
      int nrow = wn * 64 + tb * 16 + ml;
      const unsigned* hb = (const unsigned*)&Bhi[nrow * FRAG_STRIDE];
      const unsigned* lb = (const unsigned*)&Blo[nrow * FRAG_STRIDE];
#pragma unroll
      for (int v = 0; v < 8; ++v) {
        int k0 = ((v & 3) << 1) + ((v >> 2) << 4) + (h << 3);
        b_hi.u[v] = hb[k0 >> 1];
        b_lo.u[v] = lb[k0 >> 1];
      }
#pragma unroll
      for (int s = 0; s < 2; ++s) {   // f16x3: hi*hi + hi*lo + lo*hi
        acc[s][tb] = __builtin_amdgcn_wmma_f32_16x16x32_f16(false, a_hi[s].v, false, b_hi.v,
                                                            (short)0, acc[s][tb], false, false);
        acc[s][tb] = __builtin_amdgcn_wmma_f32_16x16x32_f16(false, a_hi[s].v, false, b_lo.v,
                                                            (short)0, acc[s][tb], false, false);
        acc[s][tb] = __builtin_amdgcn_wmma_f32_16x16x32_f16(false, a_lo[s].v, false, b_hi.v,
                                                            (short)0, acc[s][tb], false, false);
      }
    }
    __syncthreads();                 // frag tiles / fp32 buffers free for reuse
  }

  // ---- ArcFace margin epilogue (s=32, m=0.5) ----
  const float COSM = 0.8775825618903728f;    // cos(m)
  const float SINM = 0.479425538604203f;     // sin(m)
  const float TH   = -0.8775825618903728f;   // cos(pi - m)
  const float MM   = 0.23971276930210156f;   // sin(pi - m) * m
#pragma unroll
  for (int s = 0; s < 2; ++s) {
#pragma unroll
    for (int tb = 0; tb < 4; ++tb) {
      int n_loc = wn * 64 + tb * 16 + ml;
      int gn = n0 + n_loc;
      float rw_ = rnW[n_loc];
      bool ok = (gn < C_N);
#pragma unroll
      for (int v = 0; v < 8; ++v) {
        int m_loc = wm * 32 + s * 16 + v + (h << 3);
        int gm = m0 + m_loc;
        float cosv = acc[s][tb][v] * rnA[m_loc] * rw_;
        float sine = sqrtf(fmaxf(1.0f - cosv * cosv, 0.0f));
        float phi  = cosv * COSM - sine * SINM;
        phi = (cosv - TH > 0.0f) ? phi : (cosv - MM);
        float res = ((lab[m_loc] == gn) ? phi : cosv) * 32.0f;
        if (ok) out[(size_t)gm * C_N + gn] = res;
      }
    }
  }
}

// ---------- host launcher ----------
extern "C" void kernel_launch(void* const* d_in, const int* in_sizes, int n_in,
                              void* d_out, int out_size, void* d_ws, size_t ws_size,
                              hipStream_t stream) {
  (void)in_sizes; (void)n_in; (void)out_size; (void)ws_size;
  const float*     x     = (const float*)d_in[0];
  const float*     wgt   = (const float*)d_in[1];
  const long long* label = (const long long*)d_in[2];
  float* out = (float*)d_out;

  float* rX = (float*)d_ws;      // 512 floats
  float* rW = rX + B_M;          // 84281 floats (~339 KB total ws use)

  rnorm_kernel<<<(B_M + 7) / 8, 256, 0, stream>>>(x, rX, B_M);
  rnorm_kernel<<<(C_N + 7) / 8, 256, 0, stream>>>(wgt, rW, C_N);

  dim3 grid((C_N + BN - 1) / BN, B_M / BM);   // 659 x 4
  arc_gemm<<<grid, 256, 0, stream>>>(x, wgt, label, rX, rW, out);
}
